// NT_XentWithMoreNegatives_89240830476383
// MI455X (gfx1250) — matrix operand order repaired and checked by reading, not compiled, not run on hardware
//
#include <hip/hip_runtime.h>

// NT-Xent (k=4 views) fused kernels for gfx1250 (MI455X), wave32 + WMMA +
// CDNA5 async global->LDS staging (ASYNCcnt).
//
//  Stage 1: L2-normalize all 8192 rows (f32 -> f16 workspace).
//  Stage 2: sim = Zn[0:4096] x ZnAll^T / T via v_wmma_f32_16x16x32_f16.
//           Workgroup = 8 waves x 16 rows = 128-row strip; B tiles staged
//           once per workgroup in double-buffered LDS via
//           global_load_async_to_lds_b128 (prefetch next tile while WMMAs
//           consume current from LDS). Logits bounded in [-2,2] => plain
//           sum-of-exp, no running max. Per-(row, col-chunk) partials out.
//  Stage 3: deterministic single-block finalize: combine partials, log,
//           positive logit, tree-reduce -> d_out[0] = loss.

typedef _Float16 v16h __attribute__((ext_vector_type(16)));
typedef _Float16 v8h  __attribute__((ext_vector_type(8)));
typedef _Float16 v4h  __attribute__((ext_vector_type(4)));
typedef float    v8f  __attribute__((ext_vector_type(8)));

#define NPV    2048              // N per view
#define DIMS   128               // feature dim
#define ZROWS  4096              // 2N rows of z
#define ZAROWS 8192              // 4N rows of z_all
#define CTILES (ZAROWS / 16)     // 512 column tiles
#define CSPLIT 8                 // column chunks (grid.y)
#define TPB    (CTILES / CSPLIT) // 64 tiles per block
#define WAVES  8                 // waves per workgroup (256 threads)

// ---- CDNA5 async global->LDS copy (ASYNCcnt) ------------------------------
__device__ __forceinline__ void async_copy_b128(unsigned lds_off, const void* gptr) {
  // LDS[lds_off .. +15] = MEM[gptr .. +15], tracked by ASYNCcnt
  asm volatile("global_load_async_to_lds_b128 %0, %1, off"
               :: "v"(lds_off), "v"(gptr) : "memory");
}
__device__ __forceinline__ void wait_async0() {
  asm volatile("s_wait_asynccnt 0" ::: "memory");
}

// ---------------------------------------------------------------------------
// Stage 1: one wave per row; sum(x^2) via xor-shuffle; store normalized f16.
// ---------------------------------------------------------------------------
__global__ __launch_bounds__(256) void k_normalize(const float* __restrict__ a0,
                                                   const float* __restrict__ a1,
                                                   const float* __restrict__ a2,
                                                   const float* __restrict__ a3,
                                                   _Float16* __restrict__ zn) {
  const int gtid = blockIdx.x * 256 + threadIdx.x;
  const int wid  = gtid >> 5;                 // global wave id == row id
  const int lane = threadIdx.x & 31;

  const int b = wid >> 11, r = wid & (NPV - 1);
  const float* base = (b == 0) ? a0 : (b == 1) ? a1 : (b == 2) ? a2 : a3;
  const float* src  = base + (size_t)r * DIMS;

  float4 v = ((const float4*)src)[lane];
  float ss = v.x * v.x + v.y * v.y + v.z * v.z + v.w * v.w;
#pragma unroll
  for (int off = 16; off >= 1; off >>= 1) ss += __shfl_xor(ss, off, 32);
  const float inv = 1.0f / fmaxf(sqrtf(ss), 1e-8f);

  v4h h;
  h[0] = (_Float16)(v.x * inv);
  h[1] = (_Float16)(v.y * inv);
  h[2] = (_Float16)(v.z * inv);
  h[3] = (_Float16)(v.w * inv);
  ((v4h*)(zn + (size_t)wid * DIMS))[lane] = h;
}

// ---------------------------------------------------------------------------
// Stage 2: WMMA GEMM + masked sum-of-exp, LDS-staged B with async prefetch.
// grid = (32, 8): x = 128-row strip, y = 64-column-tile chunk.
// ---------------------------------------------------------------------------
__global__ __launch_bounds__(256) void k_ntxent(const _Float16* __restrict__ zn,
                                                float* __restrict__ partial) {
  __shared__ _Float16 bbuf[2][16 * DIMS];     // double-buffered B tile (2 x 4KB)

  const int lane  = threadIdx.x & 31;
  const int wave  = threadIdx.x >> 5;
  const int arow  = lane & 15;
  const int kg    = (lane >> 4) & 1;          // lane half
  const int r0    = blockIdx.x * (WAVES * 16) + wave * 16;  // this wave's rows
  const int cbase = blockIdx.y * TPB;         // first column tile

  // --- A fragments: 16x32 f16, interleaved K layout per ISA table ---------
  // lanes 0-15 : M=lane, K = {0..7,16..23}; lanes 16-31: M=lane-16, K = {8..15,24..31}
  v16h afrag[4];
#pragma unroll
  for (int kk = 0; kk < 4; ++kk) {
    const _Float16* p = zn + (size_t)(r0 + arow) * DIMS + kk * 32 + 8 * kg;
    v8h lo = *(const v8h*)(p);
    v8h hi = *(const v8h*)(p + 16);
    v16h f;
#pragma unroll
    for (int j = 0; j < 8; ++j) { f[j] = lo[j]; f[j + 8] = hi[j]; }
    afrag[kk] = f;
  }

  // Prologue: stage column tile 0 into bbuf[0] (256 threads x 16B = 4KB).
  {
    const _Float16* g = zn + (size_t)(cbase * 16) * DIMS + threadIdx.x * 8;
    async_copy_b128((unsigned)(size_t)&bbuf[0][threadIdx.x * 8], g);
    wait_async0();
  }
  __syncthreads();

  float ssum[8] = {0.f, 0.f, 0.f, 0.f, 0.f, 0.f, 0.f, 0.f};

  for (int i = 0; i < TPB; ++i) {
    const int cur = i & 1;
    // Prefetch next B tile into the other buffer (overlaps with WMMA below).
    if (i + 1 < TPB) {
      const _Float16* g = zn + (size_t)((cbase + i + 1) * 16) * DIMS + threadIdx.x * 8;
      async_copy_b128((unsigned)(size_t)&bbuf[cur ^ 1][threadIdx.x * 8], g);
    }

    // B fragments from LDS: 32x16 f16, interleaved K per lane (matches A)
    // (lane n<16: N=n, K={0..7,16..23}; lanes 16-31: N=lane-16, K={8..15,24..31}).
    const _Float16* bp = &bbuf[cur][arow * DIMS + kg * 8];
    v8f c = {0.f, 0.f, 0.f, 0.f, 0.f, 0.f, 0.f, 0.f};
#pragma unroll
    for (int kk = 0; kk < 4; ++kk) {
      v8h blo = *(const v8h*)(bp + kk * 32);
      v8h bhi = *(const v8h*)(bp + kk * 32 + 16);
      v16h bfrag;
#pragma unroll
      for (int j = 0; j < 8; ++j) { bfrag[j] = blo[j]; bfrag[j + 8] = bhi[j]; }
      c = __builtin_amdgcn_wmma_f32_16x16x32_f16(
          /*neg_a=*/false, afrag[kk], /*neg_b=*/false, bfrag,
          /*c_mod=*/(short)0, c, /*reuse_a=*/false, /*reuse_b=*/false);
    }

    const int col = (cbase + i) * 16 + arow;  // global column in [0, 8192)
#pragma unroll
    for (int j = 0; j < 8; ++j) {
      const int r = r0 + 8 * kg + j;          // global row in [0, 4096)
      const float x = c[j] * 2.0f;            // / TEMPERATURE (0.5)
      // mask: diagonal of every N-block except the target block (1-half)
      const bool masked = ((col & (NPV - 1)) == (r & (NPV - 1))) &&
                          ((col >> 11) != (1 - (r >> 11)));
      ssum[j] += masked ? 0.0f : __expf(x);   // x in [-2,2]: no max shift
    }

    wait_async0();      // next tile fully landed in LDS
    __syncthreads();    // all waves done reading bbuf[cur] before reuse
  }

  // Reduce sum-of-exp across the 16 lanes of each C half (xor 8/4/2/1).
#pragma unroll
  for (int j = 0; j < 8; ++j) {
#pragma unroll
    for (int off = 8; off >= 1; off >>= 1) ssum[j] += __shfl_xor(ssum[j], off, 32);
  }

  // Lanes 0 / 16 hold rows r0+8*kg .. +7: write per-(row, chunk) partials.
  if ((lane & 15) == 0) {
    const int rw = r0 + 8 * kg;
#pragma unroll
    for (int j = 0; j < 8; ++j)
      partial[(size_t)(rw + j) * CSPLIT + blockIdx.y] = ssum[j];
  }
}

// ---------------------------------------------------------------------------
// Stage 3: deterministic finalize: per-row lse - pos, tree-reduce to scalar.
// ---------------------------------------------------------------------------
__global__ __launch_bounds__(256) void k_finalize(const _Float16* __restrict__ zn,
                                                  const float* __restrict__ partial,
                                                  float* __restrict__ out) {
  __shared__ float s[256];
  float acc = 0.f;
  for (int r = threadIdx.x; r < ZROWS; r += 256) {
    float tot = 0.f;
#pragma unroll
    for (int cp = 0; cp < CSPLIT; ++cp) tot += partial[(size_t)r * CSPLIT + cp];
    const float lse = __logf(tot);            // logsumexp (masked cols excluded)

    const int half = r >> 11, idx = r & (NPV - 1);
    const int tgt  = (1 - half) * NPV + idx;
    const _Float16* pa = zn + (size_t)r * DIMS;
    const _Float16* pb = zn + (size_t)tgt * DIMS;
    float dot = 0.f;
#pragma unroll 8
    for (int k = 0; k < DIMS; ++k) dot += (float)pa[k] * (float)pb[k];

    acc += lse - dot * 2.0f;
  }
  s[threadIdx.x] = acc;
  __syncthreads();
#pragma unroll
  for (int off = 128; off >= 1; off >>= 1) {
    if ((int)threadIdx.x < off) s[threadIdx.x] += s[threadIdx.x + off];
    __syncthreads();
  }
  if (threadIdx.x == 0) out[0] = s[0] * (1.0f / (float)ZROWS);
}

// ---------------------------------------------------------------------------
extern "C" void kernel_launch(void* const* d_in, const int* in_sizes, int n_in,
                              void* d_out, int out_size, void* d_ws, size_t ws_size,
                              hipStream_t stream) {
  const float* a0 = (const float*)d_in[0];
  const float* a1 = (const float*)d_in[1];
  const float* a2 = (const float*)d_in[2];
  const float* a3 = (const float*)d_in[3];

  _Float16* zn      = (_Float16*)d_ws;                                    // 2 MB
  float*    partial = (float*)((char*)d_ws +
                               (size_t)ZAROWS * DIMS * sizeof(_Float16)); // 128 KB

  // Stage 1: 8192 waves -> 1024 blocks of 256 threads.
  k_normalize<<<(ZAROWS * 32) / 256, 256, 0, stream>>>(a0, a1, a2, a3, zn);
  // Stage 2: 32 row-strips x 8 column chunks.
  dim3 grid(ZROWS / (WAVES * 16), CSPLIT);
  k_ntxent<<<grid, 256, 0, stream>>>(zn, partial);
  // Stage 3: scalar output (fully overwritten each call).
  k_finalize<<<1, 256, 0, stream>>>(zn, partial, (float*)d_out);
}
